// GraphNetNeutral_66503273611670
// MI455X (gfx1250) — matrix operand order repaired and checked by reading, not compiled, not run on hardware
//
#include <hip/hip_runtime.h>

typedef _Float16 f16;
typedef __attribute__((ext_vector_type(8)))  _Float16 v8h;
typedef __attribute__((ext_vector_type(16))) _Float16 v16h;
typedef __attribute__((ext_vector_type(8)))  float    v8f;

// ---- problem constants ----
constexpr int kB    = 256;
constexpr int kNCH  = 60;
constexpr int kNNEU = 100;
constexpr int kNV   = 5;
constexpr int kP    = 30;
constexpr int kS    = 14;
constexpr int kT    = 8;
constexpr int kHID  = 64;
constexpr int kHID2 = 32;
constexpr int kDE   = 5;
constexpr int kDO   = 6;
constexpr int kNT   = 5;
constexpr int kNR   = kNCH * (kNCH - 1); // 3540
constexpr int kNPV  = kNCH * kNV;        // 300
constexpr int kNPN  = kNCH * kNNEU;      // 6000
constexpr int kNNODE = kB * kNCH;        // 15360

// ---------- small helpers ----------
__device__ __forceinline__ v8h ld8(const f16* p) { return *(const v8h*)p; }

__device__ __forceinline__ v16h cat16(v8h lo, v8h hi) {
  return __builtin_shufflevector(lo, hi, 0,1,2,3,4,5,6,7,8,9,10,11,12,13,14,15);
}

// Build WMMA A-operand fragment: row = 32-half padded feature row (16B aligned),
// off = (lane>=16 ? 8 : 0).  Lanes 0-15 take K {0..7,16..23}, lanes 16-31 take
// K {8..15,24..31} per the CDNA5 16-bit A-matrix layout.
__device__ __forceinline__ v16h mkA(const f16* row, int off) {
  return cat16(ld8(row + off), ld8(row + 16 + off));
}

// B-operand fragment: 16 contiguous halves per lane in the packed weight blob.
__device__ __forceinline__ v16h mkB(const f16* w, int tile, int lane) {
  const f16* p = w + ((size_t)tile * 32 + lane) * 16;
  return cat16(ld8(p), ld8(p + 8));
}

__device__ __forceinline__ v8f wmma_f16(v16h a, v16h b, v8f c) {
  return __builtin_amdgcn_wmma_f32_16x16x32_f16(false, a, false, b, (short)0, c,
                                                false, false);
}

// ---------- pack kernels ----------
__global__ void k_zero_f32(float* p, int n) {
  int i = blockIdx.x * blockDim.x + threadIdx.x;
  if (i < n) p[i] = 0.0f;
}

__global__ void k_pack_x(const float* __restrict__ x, f16* __restrict__ xt) {
  int i = blockIdx.x * blockDim.x + threadIdx.x;
  if (i >= kB * kNCH * 32) return;
  int p = i & 31, n = (i >> 5) % kNCH, b = i / (32 * kNCH);
  float v = (p < kP) ? x[(b * kP + p) * kNCH + n] : 0.0f;
  xt[i] = (f16)v;
}
__global__ void k_pack_y(const float* __restrict__ y, f16* __restrict__ yt) {
  int i = blockIdx.x * blockDim.x + threadIdx.x;
  if (i >= kB * kNV * 32) return;
  int s = i & 31, v = (i >> 5) % kNV, b = i / (32 * kNV);
  float val = (s < kS) ? y[(b * kS + s) * kNV + v] : 0.0f;
  yt[i] = (f16)val;
}
__global__ void k_pack_z(const float* __restrict__ z, f16* __restrict__ zt) {
  int i = blockIdx.x * blockDim.x + threadIdx.x;
  if (i >= kB * kNNEU * 32) return;
  int t = i & 31, n = (i >> 5) % kNNEU, b = i / (32 * kNNEU);
  float v = (t < kT) ? z[(b * kT + t) * kNNEU + n] : 0.0f;
  zt[i] = (f16)v;
}

// Pack W[dout,din] into WMMA B-operand layout: dst[nt][kt][lane][h],
// element = W[nt*16+lane%16][map(kt*32+(h<8 ? (lane/16)*8+h : 16+(lane/16)*8+h-8))].
// maptype 0: identity (K<din). maptype 1: K<30 -> K ; 32<=K,K-2<din -> K-2.
__global__ void k_pack_w(const float* __restrict__ W, f16* __restrict__ dst,
                         int dout, int din, int ktiles, int ntiles, int maptype) {
  int i = blockIdx.x * blockDim.x + threadIdx.x;
  if (i >= ntiles * ktiles * 512) return;
  int h = i & 15, lane = (i >> 4) & 31;
  int kt = (i >> 9) % ktiles, nt = (i >> 9) / ktiles;
  int col = nt * 16 + (lane & 15);
  int K = kt * 32 + ((lane >> 4) << 3) + (h & 7) + ((h >> 3) << 4);
  int d;
  if (maptype == 0) d = (K < din) ? K : -1;
  else d = (K < kP) ? K : ((K >= 32 && (K - 2) < din) ? K - 2 : -1);
  float v = (col < dout && d >= 0) ? W[col * din + d] : 0.0f;
  dst[i] = (f16)v;
}

// ---------- edge MLP (3 layers, WMMA) ----------
// ETYPE: 0 = charged-charged, 1 = charged-vertex, 2 = charged-neutral
template <int ETYPE>
__global__ void __launch_bounds__(256)
k_edge(const f16* __restrict__ xt16, const f16* __restrict__ ot16,
       const f16* __restrict__ w1, const float* __restrict__ b1,
       const f16* __restrict__ w2, const float* __restrict__ b2,
       const f16* __restrict__ w3, const float* __restrict__ b3,
       float* __restrict__ ebar, int Ne) {
  __shared__ alignas(16) f16 lds[8][1024]; // per-wave 16x64 staging
  const int b = blockIdx.x;
  const int lane = threadIdx.x & 31;
  const int wave = threadIdx.x >> 5;
  const int m16 = lane & 15;
  const int off = (lane >> 4) << 3;
  const int mb = (lane >> 4) << 3;
  const int orows = (ETYPE == 1) ? kNV : ((ETYPE == 2) ? kNNEU : kNCH);
  const f16* X = xt16 + (size_t)b * kNCH * 32;
  const f16* O = ot16 + (size_t)b * orows * 32;
  float* eb = ebar + (size_t)b * kNCH * kDE;
  f16* L = lds[wave];
  const int ntile = (Ne + 15) >> 4;

  for (int t = wave; t < ntile; t += 8) {
    int e = t * 16 + m16;
    if (e >= Ne) e = Ne - 1;
    int ra, rb;
    if (ETYPE == 0) { int r = e / 59, j = e % 59; ra = r; rb = j + (j >= r); }
    else if (ETYPE == 1) { ra = e / kNV; rb = e % kNV; }
    else { ra = e / kNNEU; rb = e % kNNEU; }

    v16h A0 = mkA(X + (size_t)ra * 32, off);
    v16h A1 = mkA(O + (size_t)rb * 32, off);

    // layer 1: K=64 (2 tiles), N=64 (4 tiles)
    v8f acc[4];
#pragma unroll
    for (int nt = 0; nt < 4; nt++) {
      v8f c = {0.f,0.f,0.f,0.f,0.f,0.f,0.f,0.f};
      c = wmma_f16(A0, mkB(w1, nt * 2 + 0, lane), c);
      c = wmma_f16(A1, mkB(w1, nt * 2 + 1, lane), c);
      acc[nt] = c;
    }
#pragma unroll
    for (int nt = 0; nt < 4; nt++) {
      int col = nt * 16 + m16;
      float bias = b1[col];
#pragma unroll
      for (int r = 0; r < 8; r++) {
        float v = acc[nt][r] + bias;
        L[(mb + r) * 64 + col] = (f16)(v > 0.f ? v : 0.f);
      }
    }

    // layer 2: K=64 (2 tiles), N=32 (2 tiles)
    v16h A20 = mkA(L + m16 * 64, off);
    v16h A21 = mkA(L + m16 * 64 + 32, off);
    v8f acc2[2];
#pragma unroll
    for (int nt = 0; nt < 2; nt++) {
      v8f c = {0.f,0.f,0.f,0.f,0.f,0.f,0.f,0.f};
      c = wmma_f16(A20, mkB(w2, nt * 2 + 0, lane), c);
      c = wmma_f16(A21, mkB(w2, nt * 2 + 1, lane), c);
      acc2[nt] = c;
    }
#pragma unroll
    for (int nt = 0; nt < 2; nt++) {
      int col = nt * 16 + m16;
      float bias = b2[col];
#pragma unroll
      for (int r = 0; r < 8; r++) {
        float v = acc2[nt][r] + bias;
        L[(mb + r) * 32 + col] = (f16)(v > 0.f ? v : 0.f);
      }
    }

    // layer 3: K=32 (1 tile), N=16 (cols >= 5 are zero-weight)
    v16h A3 = mkA(L + m16 * 32, off);
    v8f c3 = {0.f,0.f,0.f,0.f,0.f,0.f,0.f,0.f};
    c3 = wmma_f16(A3, mkB(w3, 0, lane), c3);

    if (m16 < kDE) {
      float bias = b3[m16];
#pragma unroll
      for (int r = 0; r < 8; r++) {
        int ee = t * 16 + mb + r;
        if (ee < Ne) {
          float v = c3[r] + bias;
          v = v > 0.f ? v : 0.f;
          int recv = (ETYPE == 0) ? ee / 59 : ((ETYPE == 1) ? ee / kNV : ee / kNNEU);
          atomicAdd(&eb[recv * kDE + m16], v);
        }
      }
    }
  }
}

// ---------- node feature pack: [x(30) pad2 | pp5 | pv5 | pn5 | pad] ----------
__global__ void k_pack_cfeat(const float* __restrict__ x,
                             const float* __restrict__ epp,
                             const float* __restrict__ epv,
                             const float* __restrict__ epn,
                             f16* __restrict__ cf) {
  int i = blockIdx.x * blockDim.x + threadIdx.x;
  if (i >= kNNODE * 64) return;
  int k = i & 63, node = i >> 6;
  int b = node / kNCH, n = node % kNCH;
  float v = 0.0f;
  if (k < kP) v = x[(b * kP + k) * kNCH + n];
  else if (k >= 32 && k < 32 + kDE) v = epp[(size_t)node * kDE + (k - 32)];
  else if (k >= 37 && k < 37 + kDE) v = epv[(size_t)node * kDE + (k - 37)];
  else if (k >= 42 && k < 42 + kDE) v = epn[(size_t)node * kDE + (k - 42)];
  cf[i] = (f16)v;
}

// ---------- node MLP + per-batch reduction ----------
__global__ void __launch_bounds__(256)
k_node(const f16* __restrict__ cf,
       const f16* __restrict__ w1, const float* __restrict__ b1,
       const f16* __restrict__ w2, const float* __restrict__ b2,
       const f16* __restrict__ w3, const float* __restrict__ b3,
       float* __restrict__ nsum) {
  __shared__ alignas(16) f16 lds[8][1024];
  const int lane = threadIdx.x & 31;
  const int wave = threadIdx.x >> 5;
  const int m16 = lane & 15;
  const int off = (lane >> 4) << 3;
  const int mb = (lane >> 4) << 3;
  const int gw = blockIdx.x * 8 + wave;
  const int gstep = gridDim.x * 8;
  const int ntile = kNNODE / 16; // 960
  f16* L = lds[wave];

  for (int t = gw; t < ntile; t += gstep) {
    int row = t * 16 + m16;
    const f16* Arow = cf + (size_t)row * 64;
    v16h A0 = mkA(Arow, off);
    v16h A1 = mkA(Arow + 32, off);

    v8f acc[4];
#pragma unroll
    for (int nt = 0; nt < 4; nt++) {
      v8f c = {0.f,0.f,0.f,0.f,0.f,0.f,0.f,0.f};
      c = wmma_f16(A0, mkB(w1, nt * 2 + 0, lane), c);
      c = wmma_f16(A1, mkB(w1, nt * 2 + 1, lane), c);
      acc[nt] = c;
    }
#pragma unroll
    for (int nt = 0; nt < 4; nt++) {
      int col = nt * 16 + m16;
      float bias = b1[col];
#pragma unroll
      for (int r = 0; r < 8; r++) {
        float v = acc[nt][r] + bias;
        L[(mb + r) * 64 + col] = (f16)(v > 0.f ? v : 0.f);
      }
    }

    v16h A20 = mkA(L + m16 * 64, off);
    v16h A21 = mkA(L + m16 * 64 + 32, off);
    v8f acc2[2];
#pragma unroll
    for (int nt = 0; nt < 2; nt++) {
      v8f c = {0.f,0.f,0.f,0.f,0.f,0.f,0.f,0.f};
      c = wmma_f16(A20, mkB(w2, nt * 2 + 0, lane), c);
      c = wmma_f16(A21, mkB(w2, nt * 2 + 1, lane), c);
      acc2[nt] = c;
    }
#pragma unroll
    for (int nt = 0; nt < 2; nt++) {
      int col = nt * 16 + m16;
      float bias = b2[col];
#pragma unroll
      for (int r = 0; r < 8; r++) {
        float v = acc2[nt][r] + bias;
        L[(mb + r) * 32 + col] = (f16)(v > 0.f ? v : 0.f);
      }
    }

    v16h A3 = mkA(L + m16 * 32, off);
    v8f c3 = {0.f,0.f,0.f,0.f,0.f,0.f,0.f,0.f};
    c3 = wmma_f16(A3, mkB(w3, 0, lane), c3);

    if (m16 < kDO) {
      float bias = b3[m16];
#pragma unroll
      for (int r = 0; r < 8; r++) {
        int rrow = t * 16 + mb + r;
        if (rrow < kNNODE) {
          float v = c3[r] + bias;
          v = v > 0.f ? v : 0.f;
          int bb = rrow / kNCH;
          atomicAdd(&nsum[bb * kDO + m16], v);
        }
      }
    }
  }
}

// ---------- final linear: out[b] = Nsum[b] @ fc_W.T + fc_b ----------
__global__ void k_final(const float* __restrict__ nsum,
                        const float* __restrict__ fcW,
                        const float* __restrict__ fcb,
                        float* __restrict__ out) {
  int b = blockIdx.x;
  int t = threadIdx.x;
  if (t < kNT) {
    float s = fcb[t];
#pragma unroll
    for (int d = 0; d < kDO; d++) s += nsum[b * kDO + d] * fcW[t * kDO + d];
    out[b * kNT + t] = s;
  }
}

// ---------- host ----------
extern "C" void kernel_launch(void* const* d_in, const int* in_sizes, int n_in,
                              void* d_out, int out_size, void* d_ws, size_t ws_size,
                              hipStream_t stream) {
  (void)in_sizes; (void)n_in; (void)out_size; (void)ws_size;
  const float* x = (const float*)d_in[0];
  const float* y = (const float*)d_in[1];
  const float* z = (const float*)d_in[2];
  const float* fr1_W = (const float*)d_in[3];  const float* fr1_b = (const float*)d_in[4];
  const float* fr2_W = (const float*)d_in[5];  const float* fr2_b = (const float*)d_in[6];
  const float* fr3_W = (const float*)d_in[7];  const float* fr3_b = (const float*)d_in[8];
  const float* fv1_W = (const float*)d_in[9];  const float* fv1_b = (const float*)d_in[10];
  const float* fv2_W = (const float*)d_in[11]; const float* fv2_b = (const float*)d_in[12];
  const float* fv3_W = (const float*)d_in[13]; const float* fv3_b = (const float*)d_in[14];
  const float* fn1_W = (const float*)d_in[15]; const float* fn1_b = (const float*)d_in[16];
  const float* fn2_W = (const float*)d_in[17]; const float* fn2_b = (const float*)d_in[18];
  const float* fn3_W = (const float*)d_in[19]; const float* fn3_b = (const float*)d_in[20];
  const float* fo1_W = (const float*)d_in[21]; const float* fo1_b = (const float*)d_in[22];
  const float* fo2_W = (const float*)d_in[23]; const float* fo2_b = (const float*)d_in[24];
  const float* fo3_W = (const float*)d_in[25]; const float* fo3_b = (const float*)d_in[26];
  const float* fc_W  = (const float*)d_in[27]; const float* fc_b  = (const float*)d_in[28];
  float* out = (float*)d_out;

  // workspace carve-up (256B aligned slices)
  char* ws = (char*)d_ws;
  size_t off = 0;
  auto take = [&](size_t bytes) -> void* {
    void* p = ws + off;
    off = (off + bytes + 255) & ~(size_t)255;
    return p;
  };
  f16* xt16 = (f16*)take((size_t)kB * kNCH * 32 * 2);
  f16* yt16 = (f16*)take((size_t)kB * kNV * 32 * 2);
  f16* zt16 = (f16*)take((size_t)kB * kNNEU * 32 * 2);
  f16* wpp1 = (f16*)take(4096 * 2); f16* wpp2 = (f16*)take(2048 * 2); f16* wpp3 = (f16*)take(512 * 2);
  f16* wpv1 = (f16*)take(4096 * 2); f16* wpv2 = (f16*)take(2048 * 2); f16* wpv3 = (f16*)take(512 * 2);
  f16* wpn1 = (f16*)take(4096 * 2); f16* wpn2 = (f16*)take(2048 * 2); f16* wpn3 = (f16*)take(512 * 2);
  f16* wno1 = (f16*)take(4096 * 2); f16* wno2 = (f16*)take(2048 * 2); f16* wno3 = (f16*)take(512 * 2);
  // Ebar (3x) + Nsum in one contiguous zeroed block
  const int nEb = kB * kNCH * kDE;        // 76800
  const int nZero = 3 * nEb + kB * kDO;   // 231936
  float* zblk = (float*)take((size_t)nZero * 4);
  float* epp = zblk;
  float* epv = zblk + nEb;
  float* epn = zblk + 2 * nEb;
  float* nsum = zblk + 3 * nEb;
  f16* cfeat = (f16*)take((size_t)kNNODE * 64 * 2);

  const int TPB = 256;
  auto blocks = [](int n) { return (n + 255) / 256; };

  // phase 0: zero accumulators
  k_zero_f32<<<blocks(nZero), TPB, 0, stream>>>(zblk, nZero);

  // phase 1: pack inputs (f32 -> f16, padded rows of 32)
  k_pack_x<<<blocks(kB * kNCH * 32), TPB, 0, stream>>>(x, xt16);
  k_pack_y<<<blocks(kB * kNV * 32), TPB, 0, stream>>>(y, yt16);
  k_pack_z<<<blocks(kB * kNNEU * 32), TPB, 0, stream>>>(z, zt16);

  // phase 2: pack weights into WMMA B-operand layout
  //          (dout, din, ktiles, ntiles, maptype)
  k_pack_w<<<blocks(4096), TPB, 0, stream>>>(fr1_W, wpp1, kHID, 2 * kP, 2, 4, 1);
  k_pack_w<<<blocks(2048), TPB, 0, stream>>>(fr2_W, wpp2, kHID2, kHID, 2, 2, 0);
  k_pack_w<<<blocks(512),  TPB, 0, stream>>>(fr3_W, wpp3, kDE, kHID2, 1, 1, 0);
  k_pack_w<<<blocks(4096), TPB, 0, stream>>>(fv1_W, wpv1, kHID, kS + kP, 2, 4, 1);
  k_pack_w<<<blocks(2048), TPB, 0, stream>>>(fv2_W, wpv2, kHID2, kHID, 2, 2, 0);
  k_pack_w<<<blocks(512),  TPB, 0, stream>>>(fv3_W, wpv3, kDE, kHID2, 1, 1, 0);
  k_pack_w<<<blocks(4096), TPB, 0, stream>>>(fn1_W, wpn1, kHID, kT + kP, 2, 4, 1);
  k_pack_w<<<blocks(2048), TPB, 0, stream>>>(fn2_W, wpn2, kHID2, kHID, 2, 2, 0);
  k_pack_w<<<blocks(512),  TPB, 0, stream>>>(fn3_W, wpn3, kDE, kHID2, 1, 1, 0);
  k_pack_w<<<blocks(4096), TPB, 0, stream>>>(fo1_W, wno1, kHID, kP + 3 * kDE, 2, 4, 1);
  k_pack_w<<<blocks(2048), TPB, 0, stream>>>(fo2_W, wno2, kHID2, kHID, 2, 2, 0);
  k_pack_w<<<blocks(512),  TPB, 0, stream>>>(fo3_W, wno3, kDO, kHID2, 1, 1, 0);

  // phase 3: edge MLPs (WMMA) with fused scatter-add into Ebar
  k_edge<0><<<kB, TPB, 0, stream>>>(xt16, xt16, wpp1, fr1_b, wpp2, fr2_b, wpp3, fr3_b, epp, kNR);
  k_edge<1><<<kB, TPB, 0, stream>>>(xt16, yt16, wpv1, fv1_b, wpv2, fv2_b, wpv3, fv3_b, epv, kNPV);
  k_edge<2><<<kB, TPB, 0, stream>>>(xt16, zt16, wpn1, fn1_b, wpn2, fn2_b, wpn3, fn3_b, epn, kNPN);

  // phase 4: node features, node MLP (WMMA) fused with per-batch sum
  k_pack_cfeat<<<blocks(kNNODE * 64), TPB, 0, stream>>>(x, epp, epv, epn, cfeat);
  k_node<<<120, TPB, 0, stream>>>(cfeat, wno1, fo1_b, wno2, fo2_b, wno3, fo3_b, nsum);

  // phase 5: final linear
  k_final<<<kB, 32, 0, stream>>>(nsum, fc_W, fc_b, out);
}
